// NetModule_26963804684534
// MI455X (gfx1250) — matrix operand order repaired and checked by reading, not compiled, not run on hardware
//
#include <hip/hip_runtime.h>

// 1-D valid cross-correlation via Toeplitz-filter WMMA (gfx1250, wave32).
// out[i] = sum_j x[i+j] * w[j],  N=8388608, K=4096, out_len = N-K+1.
// B-stage global->LDS copies are double-buffered through the Tensor Data
// Mover (TDM), overlapped with WMMA compute.

#define NX   8388608
#define KF   4096
#define NOUT (NX - KF + 1)      // 8384513

#define NCHUNK 129              // k-chunks of 32 covering j in [-15, 4128)
#define SCHUNK 16               // chunks of B staged in LDS at a time
#define NSTAGE ((NCHUNK + SCHUNK - 1) / SCHUNK)   // 9
#define TPW    4                // output tiles (of 256) per wave
#define WAVES  8
#define WG_OUT (WAVES * TPW * 256)                // 8192 outputs per workgroup
#define XWIN   12304            // (WG_OUT-256) + 248 + (NCHUNK-1)*32 + 23 + 1, mult of 4

typedef _Float16     v16h  __attribute__((ext_vector_type(16)));
typedef _Float16     v8h   __attribute__((ext_vector_type(8)));
typedef float        v8f   __attribute__((ext_vector_type(8)));
typedef unsigned int u32x4 __attribute__((ext_vector_type(4)));
typedef int          i32x4 __attribute__((ext_vector_type(4)));
typedef int          i32x8 __attribute__((ext_vector_type(8)));

// ---------------------------------------------------------------------------
// Prep: bake shifted-filter (Toeplitz) tiles into WMMA B-register layout, f16.
// B element for chunk c, lane L, vector element e (0..15):
//   K = (e<8 ? 0 : 16) + (L<16 ? 0 : 8) + e%8,  n = L%16,  value = w[c*32 + K - n]  (0 outside).
// Layout (halves): bt[c*512 + part*256 + L*8 + t], e = part*8 + t, so each
// wave fetches b with two lane-linear, conflict-free ds_load_b128.
// ---------------------------------------------------------------------------
__global__ void build_btiles(const float* __restrict__ w,
                             _Float16* __restrict__ bt) {
    int gid = blockIdx.x * blockDim.x + threadIdx.x;   // 129*512 = 66048 total
    if (gid >= NCHUNK * 512) return;
    int t    = gid & 7;
    int L    = (gid >> 3) & 31;
    int part = (gid >> 8) & 1;
    int c    = gid >> 9;
    int j = c * 32 + part * 16 + ((L < 16) ? 0 : 8) + t - (L & 15);
    float v = (j >= 0 && j < KF) ? w[j] : 0.0f;
    bt[gid] = (_Float16)v;
}

// Issue a 1-D TDM copy: nelem8 8-byte units from global `src` to LDS offset
// `lds_off`. Must be executed by exactly one (fully active) wave.
__device__ __forceinline__ void tdm_load_1d(const void* src, unsigned lds_off,
                                            unsigned nelem8) {
    unsigned long long ga = (unsigned long long)src;
    u32x4 g0;
    g0[0] = 1u;                                  // count=1, user descriptor
    g0[1] = lds_off;                             // D#.lds_addr
    g0[2] = (unsigned)(ga & 0xFFFFFFFFu);        // global_addr[31:0]
    g0[3] = (unsigned)((ga >> 32) & 0x01FFFFFFu) // global_addr[56:32]
          | 0x80000000u;                         // type = 2 ("image")
    i32x8 g1;
    g1[0] = (3 << 16);                           // data_size = 3 (8 bytes)
    g1[1] = (int)((nelem8 & 0xFFFFu) << 16);     // tensor_dim0[15:0]
    g1[2] = (int)((nelem8 >> 16) | (1u << 16));  // tensor_dim0[31:16] | dim1 lo
    g1[3] = (int)((nelem8 & 0xFFFFu) << 16);     // dim1 hi=0 | tile_dim0
    g1[4] = 1;                                   // tile_dim1 = 1, tile_dim2 = 0
    g1[5] = (int)nelem8;                         // tensor_dim0_stride[31:0]
    g1[6] = 0;
    g1[7] = 0;
    i32x4 gz = {0, 0, 0, 0};
#if defined(__clang_major__) && (__clang_major__ >= 23)
    i32x8 gz8 = {0, 0, 0, 0, 0, 0, 0, 0};
    __builtin_amdgcn_tensor_load_to_lds(g0, g1, gz, gz, gz8, 0);
#else
    __builtin_amdgcn_tensor_load_to_lds(g0, g1, gz, gz, 0);
#endif
}

// ---------------------------------------------------------------------------
// Main kernel: 256 threads = 8 waves; each wave owns 4 accumulator tiles.
// ---------------------------------------------------------------------------
__global__ __launch_bounds__(256)
void corr_wmma(const float* __restrict__ x,
               const _Float16* __restrict__ bt,
               float* __restrict__ out) {
    __shared__ __align__(16) _Float16 xh[XWIN];                // x window, f16
    __shared__ __align__(16) _Float16 bstage[2][SCHUNK * 512]; // B double buf

    const int tid   = threadIdx.x;
    const int lane  = tid & 31;
    const int wv    = tid >> 5;
    const int obase = blockIdx.x * WG_OUT;

    // Kick off stage-0 B DMA immediately; it overlaps the x conversion below.
    if (wv == 0) {
        tdm_load_1d(bt, (unsigned)(uintptr_t)(void*)&bstage[0][0],
                    SCHUNK * 128u);
    }

    // ---- stage x[obase .. obase+XWIN) into LDS as f16 (vectorized) ----
    {
        const float* xp = x + obase;
        for (int i4 = tid; i4 < XWIN / 4; i4 += 256) {
            int g = obase + 4 * i4;
            float4 v;
            if (g + 3 < NX) {
                v = *(const float4*)(xp + 4 * i4);
                __builtin_prefetch(xp + 4 * i4 + 4096, 0, 1); // global_prefetch_b8
            } else {
                v.x = (g + 0 < NX) ? xp[4 * i4 + 0] : 0.0f;
                v.y = (g + 1 < NX) ? xp[4 * i4 + 1] : 0.0f;
                v.z = (g + 2 < NX) ? xp[4 * i4 + 2] : 0.0f;
                v.w = (g + 3 < NX) ? xp[4 * i4 + 3] : 0.0f;
            }
            xh[4 * i4 + 0] = (_Float16)v.x;
            xh[4 * i4 + 1] = (_Float16)v.y;
            xh[4 * i4 + 2] = (_Float16)v.z;
            xh[4 * i4 + 3] = (_Float16)v.w;
        }
    }

    // A-matrix per-lane addressing (16-bit A 16x32 layout):
    // lane L: row m = L%16, kb = (L<16?0:8); elements 0..7 at a0, 8..15 at a0+16.
    const int mrow  = lane & 15;
    const int kba   = (lane < 16) ? 0 : 8;
    const int abase = 16 * mrow + kba;

    v8f acc[TPW];
#pragma unroll
    for (int t = 0; t < TPW; ++t) acc[t] = (v8f){};

    if (wv == 0) __builtin_amdgcn_s_wait_tensorcnt(0);
    __syncthreads();   // xh + bstage[0] ready for everyone

    for (int s = 0; s < NSTAGE; ++s) {
        const int c0 = s * SCHUNK;
        const int nc = (NCHUNK - c0 < SCHUNK) ? (NCHUNK - c0) : SCHUNK;

        // prefetch next stage's B tiles into the other buffer (overlapped DMA)
        if (s + 1 < NSTAGE && wv == 0) {
            const int c1  = (s + 1) * SCHUNK;
            const int nc1 = (NCHUNK - c1 < SCHUNK) ? (NCHUNK - c1) : SCHUNK;
            tdm_load_1d(bt + (size_t)c1 * 512,
                        (unsigned)(uintptr_t)(void*)&bstage[(s + 1) & 1][0],
                        (unsigned)nc1 * 128u);
        }

        const _Float16* bs = &bstage[s & 1][0];
        for (int cc = 0; cc < nc; ++cc) {
            const int kc32 = (c0 + cc) * 32;

            // issue ALL chunk loads first (10x ds_load_b128), then 4 WMMAs
            v8h blo = *(const v8h*)(bs + cc * 512 + lane * 8);
            v8h bhi = *(const v8h*)(bs + cc * 512 + 256 + lane * 8);
            v8h alo[TPW], ahi[TPW];
#pragma unroll
            for (int t = 0; t < TPW; ++t) {
                const int ax = (wv * TPW + t) * 256 + abase + kc32;
                alo[t] = *(const v8h*)(xh + ax);
                ahi[t] = *(const v8h*)(xh + ax + 16);
            }
            v16h b = __builtin_shufflevector(blo, bhi,
                     0,1,2,3,4,5,6,7,8,9,10,11,12,13,14,15);
#pragma unroll
            for (int t = 0; t < TPW; ++t) {
                v16h a = __builtin_shufflevector(alo[t], ahi[t],
                         0,1,2,3,4,5,6,7,8,9,10,11,12,13,14,15);
                // D = A x B + C, f32 accumulate
                acc[t] = __builtin_amdgcn_wmma_f32_16x16x32_f16(
                    /*neg_a=*/false, a, /*neg_b=*/false, b,
                    /*c_mod=*/(short)0, acc[t],
                    /*reuse_a=*/false, /*reuse_b=*/false);
            }
        }

        // next buffer fully landed + everyone done reading current buffer
        if (wv == 0) __builtin_amdgcn_s_wait_tensorcnt(0);
        __syncthreads();
    }

    // ---- store: C 16x16 f32 layout (VGPR r: lanes 0-15 -> M=r, 16-31 -> M=8+r)
    const int nn = lane & 15;
    const int mb = (lane < 16) ? 0 : 8;
    if (obase + WG_OUT <= NOUT) {
        // interior block: branch-free stores
#pragma unroll
        for (int t = 0; t < TPW; ++t) {
            float* op = out + obase + (wv * TPW + t) * 256 + nn;
#pragma unroll
            for (int r = 0; r < 8; ++r)
                op[16 * (mb + r)] = acc[t][r];
        }
    } else {
        // last block: guarded tail
#pragma unroll
        for (int t = 0; t < TPW; ++t) {
            const int toff = obase + (wv * TPW + t) * 256;
#pragma unroll
            for (int r = 0; r < 8; ++r) {
                const int o = toff + 16 * (mb + r) + nn;
                if (o < NOUT) out[o] = acc[t][r];
            }
        }
    }
}

// ---------------------------------------------------------------------------
extern "C" void kernel_launch(void* const* d_in, const int* in_sizes, int n_in,
                              void* d_out, int out_size, void* d_ws, size_t ws_size,
                              hipStream_t stream) {
    const float* x = (const float*)d_in[0];   // input, N f32
    const float* w = (const float*)d_in[1];   // filter, K f32
    float* out     = (float*)d_out;           // N-K+1 f32
    _Float16* bt   = (_Float16*)d_ws;         // 129*512 f16 = 132,096 B scratch

    // 1) bake Toeplitz filter tiles in WMMA B layout
    build_btiles<<<(NCHUNK * 512 + 255) / 256, 256, 0, stream>>>(w, bt);

    // 2) main correlation
    const int nblocks = (NOUT + WG_OUT - 1) / WG_OUT;   // 1024
    corr_wmma<<<nblocks, 256, 0, stream>>>(x, bt, out);
}
